// RoPEAttention_43765716746982
// MI455X (gfx1250) — matrix operand order repaired and checked by reading, not compiled, not run on hardware
//
#include <hip/hip_runtime.h>
#include <hip/hip_bf16.h>
#include <stdint.h>

typedef __attribute__((ext_vector_type(16))) _Float16 v16h;
typedef __attribute__((ext_vector_type(8)))  float    v8f;

#define LSEQ 4096
#define EMB  256
#define HDIM 32

// ---------------------------------------------------------------------------
// Fragment helpers.  All LDS tiles are stored in "fragment order" so that a
// lane's 16 f16 fragment elements are 32 contiguous bytes -> 2x ds_load_b128.
//
// A 16x32 f16 (ISA §7.12.2): lanes 0-15 hold K 0-7,16-23; lanes 16-31 hold
// K 8-15,24-31.  This code stores each A row in plain k order, grpswap = identity
// (pos = grpswap(k>>3)*8 + (k&7)), so hi-half fragments are contiguous.
// B KxN f16: lane n reads K hi*16..hi*16+15 for its column, as A does -> store source
// as [n][k] ([row][feature]) and fragments are naturally contiguous.
// ---------------------------------------------------------------------------
__device__ __forceinline__ int grpswap(int g) {            // 0,1,2,3 -> 0,1,2,3
  return g;
}
__device__ __forceinline__ v16h ld_frag(const _Float16* p) {
  return *(const v16h*)p;                                  // 32B -> 2x ds_load_b128
}
__device__ __forceinline__ void store8_f16(_Float16* dst, float4 a, float4 b) {
  union { _Float16 h[8]; uint4 u; } t;
  t.h[0] = (_Float16)a.x; t.h[1] = (_Float16)a.y;
  t.h[2] = (_Float16)a.z; t.h[3] = (_Float16)a.w;
  t.h[4] = (_Float16)b.x; t.h[5] = (_Float16)b.y;
  t.h[6] = (_Float16)b.z; t.h[7] = (_Float16)b.w;
  *(uint4*)dst = t.u;                                      // ds_store_b128
}
__device__ __forceinline__ v8f wmma_f16(v16h a, v16h b, v8f c) {
  return __builtin_amdgcn_wmma_f32_16x16x32_f16(false, a, false, b,
                                                (short)0, c, false, false);
}

// ---------------------------------------------------------------------------
// CDNA5 async global->LDS copy (CDNA5 ISA) + wait
// ---------------------------------------------------------------------------
__device__ __forceinline__ void async_load_b128(uint32_t lds_byte_off, uint64_t gaddr) {
  asm volatile("global_load_async_to_lds_b128 %0, %1, off"
               :: "v"(lds_byte_off), "v"(gaddr) : "memory");
}
__device__ __forceinline__ void wait_asynccnt0() {
  asm volatile("s_wait_asynccnt 0" ::: "memory");
}

// ---------------------------------------------------------------------------
// Kernel 0: axial RoPE tables  cos/sin [4096 positions][16 pairs]
// ---------------------------------------------------------------------------
__global__ __launch_bounds__(256) void k_rope_tables(float* __restrict__ cosT,
                                                     float* __restrict__ sinT) {
  const int idx = blockIdx.x * 256 + threadIdx.x;
  if (idx >= LSEQ * 16) return;
  const int pos = idx >> 4;
  const int i   = idx & 15;
  const int j   = i & 7;
  const float freq = powf(10000.0f, -(float)j * 0.125f);   // theta^{-4j/32}
  const float t    = (i < 8) ? (float)(pos & 63) : (float)(pos >> 6);
  const float ang  = t * freq;
  cosT[idx] = cosf(ang);
  sinT[idx] = sinf(ang);
}

// ---------------------------------------------------------------------------
// Kernel 1: projection  Y = X @ W^T + b  (+optional RoPE), f32 in -> f16 out.
// block = 256 (8 waves), tile = 128 rows x 256 cols, K chunks of 32.
// ---------------------------------------------------------------------------
__global__ __launch_bounds__(256) void k_proj_in(
    const float* __restrict__ X, const float* __restrict__ W,
    const float* __restrict__ bias, _Float16* __restrict__ Y,
    const float* __restrict__ cosT, const float* __restrict__ sinT,
    int apply_rope) {
  __shared__ __attribute__((aligned(32))) _Float16 sX[128 * 32]; // A, frag order
  __shared__ __attribute__((aligned(32))) _Float16 sW[256 * 32]; // B: [n][k]

  const int tid  = threadIdx.x;
  const int lane = tid & 31;
  const int w    = tid >> 5;
  const int nl   = lane & 15;
  const int hi   = lane >> 4;
  const int rowbase = blockIdx.x * 128;

  v8f zero = {};
  v8f acc[16];
#pragma unroll
  for (int t = 0; t < 16; ++t) acc[t] = zero;

  for (int kc = 0; kc < 8; ++kc) {
    const int k0 = kc * 32;
    {  // stage X tile in A-fragment order (f32 -> f16, vector ld/st)
      const int r = tid >> 1;
      const int h = (tid & 1);              // which 16-wide half of the chunk
      const float* src = X + (size_t)(rowbase + r) * EMB + k0 + h * 16;
      __builtin_prefetch(src + 32, 0, 3);   // global_prefetch_b8: next chunk
      const float4 f0 = ((const float4*)src)[0];
      const float4 f1 = ((const float4*)src)[1];
      const float4 f2 = ((const float4*)src)[2];
      const float4 f3 = ((const float4*)src)[3];
      _Float16* drow = &sX[r * 32];
      const int g0 = grpswap(h * 2);
      const int g1 = grpswap(h * 2 + 1);
      store8_f16(drow + g0 * 8, f0, f1);
      store8_f16(drow + g1 * 8, f2, f3);
    }
    {  // stage W chunk as [n][k] -- direct row copy, no transpose needed
      const float* src = W + (size_t)tid * EMB + k0;
#pragma unroll
      for (int g = 0; g < 4; ++g) {
        const float4 a = ((const float4*)src)[2 * g];
        const float4 b = ((const float4*)src)[2 * g + 1];
        store8_f16(&sW[tid * 32 + g * 8], a, b);
      }
    }
    __syncthreads();
    const v16h a = ld_frag(&sX[(w * 16 + nl) * 32 + hi * 16]);
#pragma unroll
    for (int t = 0; t < 16; ++t) {
      const v16h b = ld_frag(&sW[(t * 16 + nl) * 32 + hi * 16]);
      acc[t] = wmma_f16(a, b, acc[t]);
    }
    __syncthreads();
  }

  // epilogue: bias + RoPE (pair partner is the adjacent lane) + f16 store
#pragma unroll
  for (int t = 0; t < 16; ++t) {
    const int n = t * 16 + nl;
    const int h = n >> 5;
    const int d = n & 31;
    const float bn = bias[n];
#pragma unroll
    for (int r = 0; r < 8; ++r) {
      const int row  = rowbase + w * 16 + r + 8 * hi;
      const int bidx = row >> 12;
      const int l    = row & 4095;
      float val = acc[t][r] + bn;
      if (apply_rope) {
        const int pi  = d >> 1;
        const float c = cosT[l * 16 + pi];
        const float s = sinT[l * 16 + pi];
        const float part = __shfl_xor(val, 1, 32);
        val = (d & 1) ? (c * val + s * part) : (c * val - s * part);
      }
      Y[(((size_t)(bidx * 8 + h) * LSEQ + l) * HDIM) + d] = (_Float16)val;
    }
  }
}

// ---------------------------------------------------------------------------
// Kernel 2: flash attention.  grid = (L/128, B*H), block = 256 (8 waves).
// K/V tiles (64 rows) double-buffered; K staged with async DMA while WMMAs
// run on the previous tile; V staged transposed [d][kv] for contiguous
// B fragments; P goes through a per-wave LDS slab in A-fragment order.
// ---------------------------------------------------------------------------
__global__ __launch_bounds__(256) void k_attn(
    const _Float16* __restrict__ Qh, const _Float16* __restrict__ Kh,
    const _Float16* __restrict__ Vh, _Float16* __restrict__ Oh) {
  __shared__ __attribute__((aligned(32))) _Float16 sQ[128 * 32];      // A frag order
  __shared__ __attribute__((aligned(32))) _Float16 sK[2][64 * 32];    // [kv][d]
  __shared__ __attribute__((aligned(32))) _Float16 sVt[2][32 * 64];   // [d][kv]
  __shared__ __attribute__((aligned(32))) _Float16 sP[8][16 * 64];    // A frag order

  const int tid  = threadIdx.x;
  const int lane = tid & 31;
  const int w    = tid >> 5;
  const int hi   = lane >> 4;
  const int nl   = lane & 15;
  const int bh   = blockIdx.y;
  const int qbase = blockIdx.x * 128;

  const _Float16* Qb = Qh + (size_t)bh * LSEQ * HDIM;
  const _Float16* Kb = Kh + (size_t)bh * LSEQ * HDIM;
  const _Float16* Vb = Vh + (size_t)bh * LSEQ * HDIM;

  {  // Q tile -> LDS in A-fragment order (straight uint4 copy; grpswap is identity)
    const uint4* src = (const uint4*)(Qb + (size_t)qbase * HDIM);
    uint4* dst = (uint4*)sQ;
#pragma unroll
    for (int i = 0; i < 2; ++i) {
      const int u = tid + i * 256;
      dst[(u & ~3) | grpswap(u & 3)] = src[u];
    }
  }

  const uint32_t ldsK0 = (uint32_t)(uintptr_t)&sK[0][0];
  const uint32_t ldsK1 = (uint32_t)(uintptr_t)&sK[1][0];

  // stage K (async DMA) + V (transposed) for one tile into buffer `buf`
  auto stageKV = [&](int buf, int kv0) {
    const int r  = tid >> 1;
    const int hb = (tid & 1) * 16;
    async_load_b128((buf ? ldsK1 : ldsK0) + r * 64 + hb,
                    (uint64_t)(uintptr_t)((const char*)(Kb + (size_t)(kv0 + r) * HDIM) + hb));
    const int vr = tid >> 2;
    const int db = (tid & 3) * 8;
    union { uint4 u; _Float16 h[8]; } t;
    t.u = *(const uint4*)(Vb + (size_t)(kv0 + vr) * HDIM + db);
#pragma unroll
    for (int e = 0; e < 8; ++e) sVt[buf][(db + e) * 64 + vr] = t.h[e];
  };

  stageKV(0, 0);
  __syncthreads();              // Q tile visible
  const v16h aq = ld_frag(&sQ[(w * 16 + nl) * 32 + hi * 16]);

  const float scale = 0.17677669529663687f;    // 1/sqrt(32)
  float mrow[8], lrow[8];
  v8f zero = {};
  v8f oacc[2] = {zero, zero};
#pragma unroll
  for (int r = 0; r < 8; ++r) { mrow[r] = -1e30f; lrow[r] = 0.0f; }

  for (int kt = 0; kt < 64; ++kt) {
    const int cur = kt & 1;
    wait_asynccnt0();           // this wave's async K for tile kt has landed
    __syncthreads();            // all staging visible; prev reads of buffers done
    if (kt < 63) stageKV(1 - cur, (kt + 1) * 64);   // overlap DMA with WMMA

    // S = Q K^T : four 16x16 tiles, single K-chunk (HEAD_DIM == 32)
    v8f s[4];
#pragma unroll
    for (int t = 0; t < 4; ++t) {
      const v16h bk = ld_frag(&sK[cur][(t * 16 + nl) * 32 + hi * 16]);
      s[t] = wmma_f16(aq, bk, zero);
    }

    // online softmax; row stats per accumulator register, 16-lane reductions
    float alpha[8];
#pragma unroll
    for (int r = 0; r < 8; ++r) {
      float mx = s[0][r] * scale;
#pragma unroll
      for (int t = 1; t < 4; ++t) mx = fmaxf(mx, s[t][r] * scale);
#pragma unroll
      for (int off = 1; off < 16; off <<= 1) mx = fmaxf(mx, __shfl_xor(mx, off, 32));
      const float mnew = fmaxf(mrow[r], mx);
      alpha[r] = __expf(mrow[r] - mnew);
      mrow[r]  = mnew;
    }

    _Float16* sPw = &sP[w][0];
    float ps[8];
#pragma unroll
    for (int r = 0; r < 8; ++r) ps[r] = 0.0f;
#pragma unroll
    for (int t = 0; t < 4; ++t) {
      const int c      = t >> 1;
      const int klocal = (t & 1) * 16 + nl;
      const int pos    = c * 32 + (grpswap(klocal >> 3) << 3) + (klocal & 7);
#pragma unroll
      for (int r = 0; r < 8; ++r) {
        const float p = __expf(s[t][r] * scale - mrow[r]);
        ps[r] += p;
        sPw[(r + 8 * hi) * 64 + pos] = (_Float16)p;   // A-fragment order
      }
    }
#pragma unroll
    for (int r = 0; r < 8; ++r) {
#pragma unroll
      for (int off = 1; off < 16; off <<= 1) ps[r] += __shfl_xor(ps[r], off, 32);
      lrow[r] = lrow[r] * alpha[r] + ps[r];
    }
#pragma unroll
    for (int o = 0; o < 2; ++o)
#pragma unroll
      for (int r = 0; r < 8; ++r) oacc[o][r] *= alpha[r];

    __syncthreads();            // sP writes -> A-fragment reads

    // O += P @ V : two K-chunks x two output halves, all contiguous frags
    const v16h pa0 = ld_frag(&sPw[nl * 64 + hi * 16]);
    const v16h pa1 = ld_frag(&sPw[nl * 64 + 32 + hi * 16]);
#pragma unroll
    for (int o = 0; o < 2; ++o) {
      const v16h bv0 = ld_frag(&sVt[cur][(o * 16 + nl) * 64 + hi * 16]);
      const v16h bv1 = ld_frag(&sVt[cur][(o * 16 + nl) * 64 + 32 + hi * 16]);
      oacc[o] = wmma_f16(pa0, bv0, oacc[o]);
      oacc[o] = wmma_f16(pa1, bv1, oacc[o]);
    }
  }

  // normalize, write token-major f16 [b*L + l][h*32 + d]
  const int b = bh >> 3, h = bh & 7;
#pragma unroll
  for (int o = 0; o < 2; ++o)
#pragma unroll
    for (int r = 0; r < 8; ++r) {
      const int lq = qbase + w * 16 + r + 8 * hi;
      const float val = oacc[o][r] / lrow[r];
      Oh[((size_t)(b * LSEQ + lq)) * EMB + h * 32 + o * 16 + nl] = (_Float16)val;
    }
}

// ---------------------------------------------------------------------------
// Kernel 3: output projection, f16 in -> f32 out.  out = X @ Wo^T + bo
// ---------------------------------------------------------------------------
__global__ __launch_bounds__(256) void k_proj_out(
    const _Float16* __restrict__ X, const float* __restrict__ W,
    const float* __restrict__ bias, float* __restrict__ out) {
  __shared__ __attribute__((aligned(32))) _Float16 sX[128 * 32]; // A frag order
  __shared__ __attribute__((aligned(32))) _Float16 sW[256 * 32]; // B: [n][k]

  const int tid  = threadIdx.x;
  const int lane = tid & 31;
  const int w    = tid >> 5;
  const int nl   = lane & 15;
  const int hi   = lane >> 4;
  const int rowbase = blockIdx.x * 128;

  v8f zero = {};
  v8f acc[16];
#pragma unroll
  for (int t = 0; t < 16; ++t) acc[t] = zero;

  for (int kc = 0; kc < 8; ++kc) {
    const int k0 = kc * 32;
    {  // stage f16 X tile in A-fragment order (pure uint4 moves)
      const int r = tid >> 1;
      const int h = (tid & 1);
      const uint4* src = (const uint4*)(X + (size_t)(rowbase + r) * EMB + k0 + h * 16);
      uint4* drow = (uint4*)&sX[r * 32];
      drow[grpswap(h * 2)]     = src[0];
      drow[grpswap(h * 2 + 1)] = src[1];
    }
    {
      const float* src = W + (size_t)tid * EMB + k0;
#pragma unroll
      for (int g = 0; g < 4; ++g) {
        const float4 a = ((const float4*)src)[2 * g];
        const float4 b = ((const float4*)src)[2 * g + 1];
        store8_f16(&sW[tid * 32 + g * 8], a, b);
      }
    }
    __syncthreads();
    const v16h a = ld_frag(&sX[(w * 16 + nl) * 32 + hi * 16]);
#pragma unroll
    for (int t = 0; t < 16; ++t) {
      const v16h b = ld_frag(&sW[(t * 16 + nl) * 32 + hi * 16]);
      acc[t] = wmma_f16(a, b, acc[t]);
    }
    __syncthreads();
  }

#pragma unroll
  for (int t = 0; t < 16; ++t) {
    const int n = t * 16 + nl;
    const float bn = bias[n];
#pragma unroll
    for (int r = 0; r < 8; ++r) {
      const int row = rowbase + w * 16 + r + 8 * hi;
      out[(size_t)row * EMB + n] = acc[t][r] + bn;
    }
  }
}

// ---------------------------------------------------------------------------
extern "C" void kernel_launch(void* const* d_in, const int* in_sizes, int n_in,
                              void* d_out, int out_size, void* d_ws, size_t ws_size,
                              hipStream_t stream) {
  const float* q  = (const float*)d_in[0];
  const float* k  = (const float*)d_in[1];
  const float* v  = (const float*)d_in[2];
  const float* Wq = (const float*)d_in[3];
  const float* bq = (const float*)d_in[4];
  const float* Wk = (const float*)d_in[5];
  const float* bk = (const float*)d_in[6];
  const float* Wv = (const float*)d_in[7];
  const float* bv = (const float*)d_in[8];
  const float* Wo = (const float*)d_in[9];
  const float* bo = (const float*)d_in[10];
  float* out = (float*)d_out;

  const size_t headElems = (size_t)2 * 8 * LSEQ * HDIM;   // 2,097,152 f16 = 4 MB
  char* ws = (char*)d_ws;
  float*    cosT = (float*)ws;                            // 256 KB
  float*    sinT = (float*)(ws + 262144);                 // 256 KB
  _Float16* Qh   = (_Float16*)(ws + 524288);
  _Float16* Kh   = Qh + headElems;
  _Float16* Vh   = Kh + headElems;
  _Float16* Oh   = Vh + headElems;                        // token-major [8192][256]

  k_rope_tables<<<dim3(256), dim3(256), 0, stream>>>(cosT, sinT);
  k_proj_in<<<dim3(64), dim3(256), 0, stream>>>(q, Wq, bq, Qh, cosT, sinT, 1);
  k_proj_in<<<dim3(64), dim3(256), 0, stream>>>(k, Wk, bk, Kh, cosT, sinT, 1);
  k_proj_in<<<dim3(64), dim3(256), 0, stream>>>(v, Wv, bv, Vh, cosT, sinT, 0);
  k_attn<<<dim3(32, 16), dim3(256), 0, stream>>>(Qh, Kh, Vh, Oh);
  k_proj_out<<<dim3(64), dim3(256), 0, stream>>>(Oh, Wo, bo, out);
}
